// RecurrentBlock_74594991997395
// MI455X (gfx1250) — compile-verified, matrix operand order checked by reading, not run on hardware
//
#include <hip/hip_runtime.h>
#include <stdint.h>

// ---------------------------------------------------------------------------
// RG-LRU recurrent block for MI455X (gfx1250, wave32, WMMA + async-LDS DMA).
//   0. one-time: transpose+convert all GEMM weights to bf16 [N][K] in ws
//   1. y  = gelu(x @ Wy + by)              wmma bf16 GEMM
//   2. xb = x @ Wx + bx                    wmma bf16 GEMM
//   3. conv = causal depthwise conv1d(xb)  elementwise
//   4. gate linear GEMMs (block-diag, grid.z = head)
//   5. RG-LRU elementwise -> a, norm_x
//   6. chunked linear scan (3-pass, chunk=64)
//   7. out = (h * y) @ Wo + bo             h*y fused into A staging
// GEMM: A tile f32->bf16 packed in VALU (co-executes with WMMA);
//       B tile = pre-transposed bf16 weights DMA'd straight into LDS with
//       global_load_async_to_lds_b128 (ASYNCcnt), zero VALU;
//       fragments = 2x ds_load_b128 each; 8x v_wmma_f32_16x16x32_bf16/K-tile.
// ---------------------------------------------------------------------------

typedef __attribute__((ext_vector_type(16))) __bf16        v16bf;
typedef __attribute__((ext_vector_type(8)))  float         v8f;
typedef __attribute__((ext_vector_type(4)))  unsigned int  v4u;

#define BM 128
#define BN 128
#define BK 32
#define KPAD 8   // rows padded to 80B: every 16B fragment run stays aligned

__device__ __forceinline__ unsigned short f2bf(float f) {
  unsigned int u = __float_as_uint(f);
  u += 0x7FFFu + ((u >> 16) & 1u);   // round-to-nearest-even
  return (unsigned short)(u >> 16);
}

__device__ __forceinline__ unsigned int pack2bf(float lo, float hi) {
  return (unsigned int)f2bf(lo) | ((unsigned int)f2bf(hi) << 16);
}

union BF16Frag { v16bf v; v4u q[2]; };

__device__ __forceinline__ float gelu_tanh(float x) {
  float x3 = x * x * x;
  return 0.5f * x * (1.0f + tanhf(0.79788456080286535588f * (x + 0.044715f * x3)));
}

// CDNA5 async global->LDS copy (16B), tracked by ASYNCcnt.
__device__ __forceinline__ void async_ld_b128(unsigned lds_off, const void* gaddr) {
  asm volatile("global_load_async_to_lds_b128 %0, %1, off"
               :: "v"(lds_off), "v"(gaddr) : "memory");
}
__device__ __forceinline__ void wait_async0() {
  asm volatile("s_wait_asynccnt 0x0" ::: "memory");
}

// C[M,N] = act( (A .* A2?) @ W + bias ); A,A2,bias,C row-major f32;
// WT = bf16 weights pre-transposed to [N][K]. grid.z strides for block-diag.
__global__ __launch_bounds__(256) void wmma_gemm_kernel(
    const float* __restrict__ A, const float* __restrict__ A2,
    const unsigned short* __restrict__ WT, const float* __restrict__ bias,
    float* __restrict__ C,
    int K, int lda, int ldwt, int ldc,
    long long a_zoff, long long wt_zoff, long long b_zoff, long long c_zoff,
    int act)
{
  // A tile row-major [m][k]; B tile [n][k] (same layout as WT rows).
  __shared__ __align__(16) unsigned short As[2][BM][BK + KPAD];
  __shared__ __align__(16) unsigned short BsT[2][BN][BK + KPAD];

  A += (long long)blockIdx.z * a_zoff;
  if (A2) A2 += (long long)blockIdx.z * a_zoff;
  WT += (long long)blockIdx.z * wt_zoff;
  if (bias) bias += (long long)blockIdx.z * b_zoff;
  C += (long long)blockIdx.z * c_zoff;

  const int tid   = threadIdx.x;
  const int lane  = tid & 31;
  const int wave  = tid >> 5;
  const int wm    = wave >> 1;   // 0..3
  const int wn    = wave & 1;    // 0..1
  const int lrow  = lane & 15;
  const int khalf = lane >> 4;

  const int rowBase = blockIdx.y * BM;
  const int colBase = blockIdx.x * BN;

  // A staging: thread packs 16 contiguous f32 of one row -> 2x b128 LDS store
  const int ar = tid >> 1;          // 0..127
  const int ac = (tid & 1) << 4;    // 0 / 16
  // B staging: thread DMA-copies 32B of one WT row (2x async b128)
  const int bn  = tid >> 1;         // 0..127 (row n)
  const int bc8 = (tid & 1) << 4;   // short offset 0 / 16

  v8f acc[2][4] = {};
  const int KT = K / BK;

#define STAGE_TILE(buf, k0)                                                   \
  {                                                                           \
    const float* s = A + (size_t)(rowBase + ar) * lda + (k0) + ac;            \
    v4u q0, q1;                                                               \
    if (A2) {                                                                 \
      const float* s2 = A2 + (size_t)(rowBase + ar) * lda + (k0) + ac;        \
      q0[0] = pack2bf(s[0] * s2[0], s[1] * s2[1]);                            \
      q0[1] = pack2bf(s[2] * s2[2], s[3] * s2[3]);                            \
      q0[2] = pack2bf(s[4] * s2[4], s[5] * s2[5]);                            \
      q0[3] = pack2bf(s[6] * s2[6], s[7] * s2[7]);                            \
      q1[0] = pack2bf(s[8] * s2[8], s[9] * s2[9]);                            \
      q1[1] = pack2bf(s[10] * s2[10], s[11] * s2[11]);                        \
      q1[2] = pack2bf(s[12] * s2[12], s[13] * s2[13]);                        \
      q1[3] = pack2bf(s[14] * s2[14], s[15] * s2[15]);                        \
    } else {                                                                  \
      q0[0] = pack2bf(s[0], s[1]);   q0[1] = pack2bf(s[2], s[3]);             \
      q0[2] = pack2bf(s[4], s[5]);   q0[3] = pack2bf(s[6], s[7]);             \
      q1[0] = pack2bf(s[8], s[9]);   q1[1] = pack2bf(s[10], s[11]);           \
      q1[2] = pack2bf(s[12], s[13]); q1[3] = pack2bf(s[14], s[15]);           \
    }                                                                         \
    *(v4u*)&As[buf][ar][ac]     = q0;                                         \
    *(v4u*)&As[buf][ar][ac + 8] = q1;                                         \
    const unsigned short* wt =                                                \
        WT + (size_t)(colBase + bn) * ldwt + (k0) + bc8;                      \
    async_ld_b128((unsigned)(uintptr_t)&BsT[buf][bn][bc8],     wt);           \
    async_ld_b128((unsigned)(uintptr_t)&BsT[buf][bn][bc8 + 8], wt + 8);       \
  }

  STAGE_TILE(0, 0);
  wait_async0();
  __syncthreads();

  for (int kt = 0; kt < KT; ++kt) {
    const int cur = kt & 1;
    if (kt + 1 < KT) {
      STAGE_TILE(cur ^ 1, (kt + 1) * BK);
    }
    if (kt + 2 < KT) {  // prefetch A tile after next (weights are L2-resident)
      __builtin_prefetch(A + (size_t)(rowBase + ar) * lda + (kt + 2) * BK + ac, 0, 1);
    }

    // fragments: per ISA 7.12.2 a lane holds K runs {khalf*8..+7, 16+khalf*8..+7}
    v16bf afrag[2];
#pragma unroll
    for (int mi = 0; mi < 2; ++mi) {
      BF16Frag f;
      const unsigned short* rp = &As[cur][wm * 32 + mi * 16 + lrow][0];
      f.q[0] = *(const v4u*)(rp + (khalf << 3));
      f.q[1] = *(const v4u*)(rp + 16 + (khalf << 3));
      afrag[mi] = f.v;
    }
    v16bf bfrag[4];
#pragma unroll
    for (int ni = 0; ni < 4; ++ni) {
      BF16Frag f;
      const unsigned short* rp = &BsT[cur][wn * 64 + ni * 16 + lrow][0];
      f.q[0] = *(const v4u*)(rp + (khalf << 3));
      f.q[1] = *(const v4u*)(rp + 16 + (khalf << 3));
      bfrag[ni] = f.v;
    }
#pragma unroll
    for (int mi = 0; mi < 2; ++mi)
#pragma unroll
      for (int ni = 0; ni < 4; ++ni)
        acc[mi][ni] = __builtin_amdgcn_wmma_f32_16x16x32_bf16(
            false, afrag[mi], false, bfrag[ni], (short)0, acc[mi][ni],
            false, false);

    wait_async0();       // next tile's DMA done before releasing the barrier
    __syncthreads();
  }
#undef STAGE_TILE

  // epilogue: f32 C/D layout — VGPR v: lanes<16 -> M=v, lanes>=16 -> M=v+8
#pragma unroll
  for (int mi = 0; mi < 2; ++mi) {
#pragma unroll
    for (int ni = 0; ni < 4; ++ni) {
      const int mbase = rowBase + wm * 32 + mi * 16 + (khalf << 3);
      const int n     = colBase + wn * 64 + ni * 16 + lrow;
      const float bv  = bias ? bias[n] : 0.0f;
#pragma unroll
      for (int v = 0; v < 8; ++v) {
        float val = acc[mi][ni][v] + bv;
        if (act == 1) val = gelu_tanh(val);
        C[(size_t)(mbase + v) * ldc + n] = val;
      }
    }
  }
}

// one-time: WT[n][k] = bf16(W[k][n]); output-coalesced. grid.z = head strides.
__global__ void transpose_bf16_kernel(const float* __restrict__ W,
                                      unsigned short* __restrict__ WT,
                                      int K, int N,
                                      long long zoff_in, long long zoff_out,
                                      long long total)
{
  const float* Wz = W + (long long)blockIdx.z * zoff_in;
  unsigned short* Oz = WT + (long long)blockIdx.z * zoff_out;
  long long idx = (long long)blockIdx.x * blockDim.x + threadIdx.x;
  if (idx >= total) return;
  int k = (int)(idx % K);
  long long n = idx / K;
  Oz[idx] = f2bf(Wz[(size_t)k * N + n]);
}

// causal depthwise conv1d over T (left zero pad, w[TW-1] hits current token)
__global__ void conv1d_kernel(const float* __restrict__ xb,
                              const float* __restrict__ cw,
                              const float* __restrict__ cb,
                              float* __restrict__ out,
                              int T, int D, long long total)
{
  long long idx = (long long)blockIdx.x * blockDim.x + threadIdx.x;
  if (idx >= total) return;
  int d = (int)(idx % D);
  long long bt = idx / D;
  int t = (int)(bt % (long long)T);
  float acc = cb[d];
#pragma unroll
  for (int i = 0; i < 4; ++i) {
    int tt = t - 3 + i;
    if (tt >= 0) acc = fmaf(cw[i * D + d], xb[idx + (long long)(tt - t) * D], acc);
  }
  out[idx] = acc;
}

// RG-LRU elementwise: gx buffer (gate_x linear -> norm_x), ga (gate_a -> a)
__global__ void rglru_elem_kernel(const float* __restrict__ conv,
                                  float* gx, float* ga,
                                  const float* __restrict__ a_param,
                                  const int* __restrict__ segpos,
                                  int T, int D, long long total)
{
  long long idx = (long long)blockIdx.x * blockDim.x + threadIdx.x;
  if (idx >= total) return;
  int d = (int)(idx % D);
  long long bt = idx / D;
  float gxv = 1.0f / (1.0f + expf(-gx[idx]));
  float gav = 1.0f / (1.0f + expf(-ga[idx]));
  float sp  = log1pf(expf(a_param[d]));        // softplus
  float log_a = -8.0f * gav * sp;              // C_CONST = 8
  float a = expf(log_a);
  float mult = sqrtf(fmaxf(1.0f - expf(2.0f * log_a), 0.0f));
  if (segpos[bt] == 0) { mult = 1.0f; a = 0.0f; }
  gx[idx] = conv[idx] * gxv * mult;            // norm_x
  ga[idx] = a;
}

// chunked scan pass 1: per-chunk carries (prod of a, local h with h_in = 0)
__global__ void scan_part1(const float* __restrict__ a,
                           const float* __restrict__ nx,
                           float* __restrict__ P, float* __restrict__ Hl,
                           int BD, int T, int D, int CH)
{
  int NC = T / CH;
  int idx = blockIdx.x * blockDim.x + threadIdx.x;
  if (idx >= BD * NC) return;
  int ch = idx % BD;
  int c  = idx / BD;
  int b = ch / D, d = ch % D;
  size_t base = ((size_t)b * T + (size_t)c * CH) * D + d;
  float h = 0.0f, p = 1.0f;
  for (int t = 0; t < CH; ++t) {
    float at = a[base + (size_t)t * D];
    h = fmaf(at, h, nx[base + (size_t)t * D]);
    p *= at;
  }
  P[(size_t)c * BD + ch]  = p;
  Hl[(size_t)c * BD + ch] = h;
}

// pass 2: tiny serial scan across chunks per channel -> chunk-entry states
__global__ void scan_part2(const float* __restrict__ P,
                           const float* __restrict__ Hl,
                           float* __restrict__ H0, int BD, int NC)
{
  int ch = blockIdx.x * blockDim.x + threadIdx.x;
  if (ch >= BD) return;
  float h = 0.0f;
  for (int c = 0; c < NC; ++c) {
    H0[(size_t)c * BD + ch] = h;
    h = fmaf(P[(size_t)c * BD + ch], h, Hl[(size_t)c * BD + ch]);
  }
}

// pass 3: replay chunks with correct entry state, write h in place over nx
__global__ void scan_part3(const float* __restrict__ a,
                           float* __restrict__ nx,
                           const float* __restrict__ H0,
                           int BD, int T, int D, int CH)
{
  int NC = T / CH;
  int idx = blockIdx.x * blockDim.x + threadIdx.x;
  if (idx >= BD * NC) return;
  int ch = idx % BD;
  int c  = idx / BD;
  int b = ch / D, d = ch % D;
  size_t base = ((size_t)b * T + (size_t)c * CH) * D + d;
  float h = H0[(size_t)c * BD + ch];
  for (int t = 0; t < CH; ++t) {
    size_t o = base + (size_t)t * D;
    h = fmaf(a[o], h, nx[o]);
    nx[o] = h;
  }
}

extern "C" void kernel_launch(void* const* d_in, const int* in_sizes, int n_in,
                              void* d_out, int out_size, void* d_ws, size_t ws_size,
                              hipStream_t stream) {
  (void)in_sizes; (void)n_in; (void)out_size; (void)ws_size;
  const int Bb = 4, T = 4096, D = 2560, H = 10, Dh = 256;
  const int M = Bb * T;                       // 16384
  const long long MD = (long long)M * D;      // 41,943,040

  const float* x      = (const float*)d_in[0];
  const int*   segp   = (const int*)  d_in[1];
  const float* Wy     = (const float*)d_in[2];
  const float* by     = (const float*)d_in[3];
  const float* Wx     = (const float*)d_in[4];
  const float* bx     = (const float*)d_in[5];
  const float* conv_w = (const float*)d_in[6];
  const float* conv_b = (const float*)d_in[7];
  const float* a_par  = (const float*)d_in[8];
  const float* ig_w   = (const float*)d_in[9];
  const float* ig_b   = (const float*)d_in[10];
  const float* ag_w   = (const float*)d_in[11];
  const float* ag_b   = (const float*)d_in[12];
  const float* Wo     = (const float*)d_in[13];
  const float* bo     = (const float*)d_in[14];

  const int CH = 64, NC = T / CH, BD = Bb * D;

  float* ws   = (float*)d_ws;
  float* y    = ws;
  float* xb   = ws + MD;
  float* conv = ws + 2 * MD;
  float* gx   = ws + 3 * MD;   // gate_x linear -> norm_x -> h
  float* ga   = ws + 4 * MD;   // gate_a linear -> a
  float* P    = ws + 5 * MD;
  float* Hl   = P + (long long)NC * BD;
  float* H0   = Hl + (long long)NC * BD;
  // bf16 transposed weights
  unsigned short* WyT = (unsigned short*)(H0 + (long long)NC * BD);
  unsigned short* WxT = WyT + (long long)D * D;
  unsigned short* WoT = WxT + (long long)D * D;
  unsigned short* igT = WoT + (long long)D * D;
  unsigned short* agT = igT + (long long)H * Dh * Dh;

  dim3 blk(256);
  dim3 gBig(D / BN, M / BM, 1);     // (20, 128)
  dim3 gHead(Dh / BN, M / BM, H);   // (2, 128, 10)

  // 0) one-time weight transpose + bf16 conversion
  long long wtot = (long long)D * D;
  transpose_bf16_kernel<<<dim3((unsigned)((wtot + 255) / 256)), blk, 0, stream>>>(
      Wy, WyT, D, D, 0, 0, wtot);
  transpose_bf16_kernel<<<dim3((unsigned)((wtot + 255) / 256)), blk, 0, stream>>>(
      Wx, WxT, D, D, 0, 0, wtot);
  transpose_bf16_kernel<<<dim3((unsigned)((wtot + 255) / 256)), blk, 0, stream>>>(
      Wo, WoT, D, D, 0, 0, wtot);
  long long htot = (long long)Dh * Dh;
  transpose_bf16_kernel<<<dim3((unsigned)((htot + 255) / 256), 1, H), blk, 0, stream>>>(
      ig_w, igT, Dh, Dh, htot, htot, htot);
  transpose_bf16_kernel<<<dim3((unsigned)((htot + 255) / 256), 1, H), blk, 0, stream>>>(
      ag_w, agT, Dh, Dh, htot, htot, htot);

  // 1) y = gelu(x @ Wy + by)
  wmma_gemm_kernel<<<gBig, blk, 0, stream>>>(x, nullptr, WyT, by, y,
      D, D, D, D, 0, 0, 0, 0, 1);
  // 2) xb = x @ Wx + bx
  wmma_gemm_kernel<<<gBig, blk, 0, stream>>>(x, nullptr, WxT, bx, xb,
      D, D, D, D, 0, 0, 0, 0, 0);
  // 3) causal depthwise conv
  long long total = MD;
  int eblocks = (int)((total + 255) / 256);
  conv1d_kernel<<<dim3(eblocks), blk, 0, stream>>>(xb, conv_w, conv_b, conv, T, D, total);
  // 4) block-diagonal gate GEMMs (grid.z = head)
  wmma_gemm_kernel<<<gHead, blk, 0, stream>>>(conv, nullptr, igT, ig_b, gx,
      Dh, D, Dh, D, (long long)Dh, htot, (long long)Dh, (long long)Dh, 0);
  wmma_gemm_kernel<<<gHead, blk, 0, stream>>>(conv, nullptr, agT, ag_b, ga,
      Dh, D, Dh, D, (long long)Dh, htot, (long long)Dh, (long long)Dh, 0);
  // 5) RG-LRU elementwise (in place over gx/ga)
  rglru_elem_kernel<<<dim3(eblocks), blk, 0, stream>>>(conv, gx, ga, a_par, segp, T, D, total);
  // 6) chunked scan: a=ga, norm_x=gx -> h written over gx
  int p1 = (BD * NC + 255) / 256;
  scan_part1<<<dim3(p1), blk, 0, stream>>>(ga, gx, P, Hl, BD, T, D, CH);
  scan_part2<<<dim3((BD + 255) / 256), blk, 0, stream>>>(P, Hl, H0, BD, NC);
  scan_part3<<<dim3(p1), blk, 0, stream>>>(ga, gx, H0, BD, T, D, CH);
  // 7) out = (h * y) @ Wo + bo   (h*y fused into the A-tile staging)
  wmma_gemm_kernel<<<gBig, blk, 0, stream>>>(gx, y, WoT, bo, (float*)d_out,
      D, D, D, D, 0, 0, 0, 0, 0);
}
